// DGCNN_87471303951062
// MI455X (gfx1250) — compile-verified, matrix operand order checked by reading, not run on hardware
//
#include <hip/hip_runtime.h>

typedef __attribute__((ext_vector_type(16))) _Float16 v16h;
typedef __attribute__((ext_vector_type(8)))  _Float16 v8h;
typedef __attribute__((ext_vector_type(8)))  float    v8f;

#define BATCH 16
#define NPTS  1024
#define KNN   20

static __device__ __forceinline__ float lrelu(float x) { return x >= 0.f ? x : 0.2f * x; }

static __device__ __forceinline__ v16h cat16(v8h lo, v8h hi) {
    return __builtin_shufflevector(lo, hi, 0,1,2,3,4,5,6,7,8,9,10,11,12,13,14,15);
}

// Load a WMMA fragment for a row-major f16 matrix.
// Per the CDNA5 16-bit A layout: lane L covers row M=L&15, K-halves
// {b..b+7} and {16+b..16+b+7} with b = 8*(L>=16): two contiguous 16B chunks.
static __device__ __forceinline__ v16h frag_row(const _Float16* __restrict__ rowp,
                                                int kb, int lane) {
    const _Float16* p = rowp + kb + ((lane >> 4) << 3);
    v8h lo = *(const v8h*)p;
    v8h hi = *(const v8h*)(p + 16);
    return cat16(lo, hi);
}

// Edge-feature chunk: e[c] = (c < C) ? xt[nbr,c]-xt[n,c] : xt[n,c-C].
// Chunks are 8-aligned and C is a multiple of 64 here, so no chunk straddles C.
static __device__ __forceinline__ v8h edge_chunk(const _Float16* __restrict__ xb,
                                                 int C, int n, int nbr, int c0) {
    if (c0 < C) {
        v8h a = *(const v8h*)(xb + nbr * C + c0);
        v8h c = *(const v8h*)(xb + n * C + c0);
        return a - c;
    }
    return *(const v8h*)(xb + n * C + (c0 - C));
}

// ---------------- BN prep: scale = g/sqrt(v+eps), shift = b - m*scale --------
__global__ void k_bnprep(const float* __restrict__ g, const float* __restrict__ bt,
                         const float* __restrict__ m, const float* __restrict__ v,
                         float* __restrict__ sc, float* __restrict__ sh, int C) {
    int c = blockIdx.x * blockDim.x + threadIdx.x;
    if (c >= C) return;
    float s = g[c] / sqrtf(v[c] + 1e-5f);
    sc[c] = s;
    sh[c] = bt[c] - m[c] * s;
}

// ---------------- float -> f16 convert --------------------------------------
__global__ void k_cvt16(const float* __restrict__ in, _Float16* __restrict__ out, int n) {
    int t = blockIdx.x * blockDim.x + threadIdx.x;
    if (t < n) out[t] = (_Float16)in[t];
}

// ---------------- x (B,3,N) -> xt (B,N,3) -----------------------------------
__global__ void k_transpose_x(const float* __restrict__ x, float* __restrict__ xt) {
    int t = blockIdx.x * blockDim.x + threadIdx.x;
    if (t >= BATCH * NPTS) return;
    int b = t >> 10, n = t & 1023;
    #pragma unroll
    for (int c = 0; c < 3; ++c)
        xt[t * 3 + c] = x[(b * 3 + c) * NPTS + n];
}

// ---------------- layer-1 pairwise distance (C=3, scalar) -------------------
__global__ void k_dist_l1(const float* __restrict__ xt, float* __restrict__ dist) {
    size_t t = (size_t)blockIdx.x * blockDim.x + threadIdx.x;
    if (t >= (size_t)BATCH * NPTS * NPTS) return;
    int m = (int)(t & 1023), n = (int)((t >> 10) & 1023), b = (int)(t >> 20);
    const float* pn = xt + (b * NPTS + n) * 3;
    const float* pm = xt + (b * NPTS + m) * 3;
    float d0 = pn[0] - pm[0], d1 = pn[1] - pm[1], d2 = pn[2] - pm[2];
    dist[t] = -(d0 * d0 + d1 * d1 + d2 * d2);
}

// ---------------- squared norms ---------------------------------------------
__global__ void k_sq(const float* __restrict__ xt, float* __restrict__ sq, int C) {
    int t = blockIdx.x * blockDim.x + threadIdx.x;
    if (t >= BATCH * NPTS) return;
    const float* p = xt + (size_t)t * C;
    float a = 0.f;
    for (int c = 0; c < C; ++c) a += p[c] * p[c];
    sq[t] = a;
}

// ---------------- WMMA Gram/distance: d = 2*A.A^T - |a|^2 - |b|^2 -----------
// One wave computes a 16n x 64m tile: A fragment loaded once per k-step and
// reused across 4 accumulators (independent accs -> no WMMA RAW hazard).
// grid: (16 m-supertiles, 64 n-tiles, 16 b), block = 32.
__global__ void k_dist_wmma(const _Float16* __restrict__ xt16, const float* __restrict__ sq,
                            float* __restrict__ dist, int C) {
    int lane = threadIdx.x;
    int b = blockIdx.z, n0 = blockIdx.y << 4, m0 = blockIdx.x << 6;
    int r15 = lane & 15, loff = (lane >> 4) << 3;
    const _Float16* xb = xt16 + (size_t)b * NPTS * C;
    const _Float16* arow = xb + (size_t)(n0 + r15) * C;
    const _Float16* brow = xb + (size_t)(m0 + r15) * C;   // subtile j adds j*16*C
    v8f acc[4] = {{}, {}, {}, {}};
    int ksteps = C >> 5;
    for (int s = 0; s < ksteps; ++s) {
        int kb = s << 5;
        v16h af = frag_row(arow, kb, lane);
        #pragma unroll
        for (int j = 0; j < 4; ++j) {
            v16h bf = frag_row(brow + (size_t)(j << 4) * C, kb, lane);
            acc[j] = __builtin_amdgcn_wmma_f32_16x16x32_f16(false, af, false, bf,
                                                            (short)0, acc[j], false, false);
        }
    }
    #pragma unroll
    for (int j = 0; j < 4; ++j) {
        int m = m0 + (j << 4) + r15;
        float sqm = sq[b * NPTS + m];
        #pragma unroll
        for (int r = 0; r < 8; ++r) {
            int n = n0 + loff + r;
            dist[((size_t)(b * NPTS + n) << 10) + m] =
                2.f * acc[j][r] - sq[b * NPTS + n] - sqm;
        }
    }
}

// ---------------- top-20 per point (register insertion) ---------------------
__global__ void k_topk(const float* __restrict__ dist, int* __restrict__ idx) {
    int t = blockIdx.x * blockDim.x + threadIdx.x;
    if (t >= BATCH * NPTS) return;
    const float* row = dist + ((size_t)t << 10);
    float vals[KNN]; int ids[KNN];
    #pragma unroll
    for (int i = 0; i < KNN; ++i) { vals[i] = -3.0e38f; ids[i] = 0; }
    for (int m = 0; m < NPTS; ++m) {
        float d = row[m];
        if (d > vals[0]) {               // vals ascending, vals[0] is current min
            int j = 0;
            #pragma unroll
            for (int s = 1; s < KNN; ++s)
                if (d > vals[s]) { vals[s - 1] = vals[s]; ids[s - 1] = ids[s]; j = s; }
            vals[j] = d; ids[j] = m;
        }
    }
    #pragma unroll
    for (int i = 0; i < KNN; ++i) idx[t * KNN + i] = ids[i];
}

// ---------------- layer-1 edgeconv (C=3, O=64, scalar) ----------------------
__global__ void k_edge_l1(const float* __restrict__ xt, const int* __restrict__ idx,
                          const float* __restrict__ w, const float* __restrict__ sc,
                          const float* __restrict__ sh, float* __restrict__ of32,
                          _Float16* __restrict__ of16, _Float16* __restrict__ cat) {
    int t = blockIdx.x * blockDim.x + threadIdx.x;
    if (t >= BATCH * NPTS * 64) return;
    int o = t & 63, n = (t >> 6) & 1023, b = t >> 16;
    int bn = b * NPTS + n;
    const float* pc = xt + bn * 3;
    float w0 = w[o*6], w1 = w[o*6+1], w2 = w[o*6+2];
    float cterm = w[o*6+3]*pc[0] + w[o*6+4]*pc[1] + w[o*6+5]*pc[2];
    float s = sc[o], h = sh[o], ymax = -3.0e38f;
    for (int k = 0; k < KNN; ++k) {
        const float* pf = xt + (b * NPTS + idx[bn * KNN + k]) * 3;
        float a = w0*(pf[0]-pc[0]) + w1*(pf[1]-pc[1]) + w2*(pf[2]-pc[2]) + cterm;
        ymax = fmaxf(ymax, lrelu(a * s + h));
    }
    of32[bn * 64 + o] = ymax;
    of16[bn * 64 + o] = (_Float16)ymax;
    cat[(size_t)bn * 512 + o] = (_Float16)ymax;
}

// ---------------- WMMA edgeconv: gather fused into A-fragment ---------------
// One wave computes a 16n x 64o tile. Per (k, k-step) the gathered edge
// fragment (loads + packed-f16 subtract) is built ONCE and reused by 4 WMMAs.
// BN + LeakyReLU + max-over-k fused in registers; prefetch next neighbor row.
// grid: (O/64, 64 n-tiles, 16 b), block = 32.
__global__ void k_edge_wmma(const _Float16* __restrict__ xt16, const int* __restrict__ idx,
                            const _Float16* __restrict__ w16, const float* __restrict__ sc,
                            const float* __restrict__ sh, float* __restrict__ of32,
                            _Float16* __restrict__ of16, _Float16* __restrict__ cat,
                            int C, int O, int coff) {
    int lane = threadIdx.x;
    int b = blockIdx.z, n0 = blockIdx.y << 4, o0 = blockIdx.x << 6;
    int r15 = lane & 15, loff = (lane >> 4) << 3;
    int twoC = C << 1, ksteps = twoC >> 5;
    const _Float16* xb = xt16 + (size_t)b * NPTS * C;
    int n = n0 + r15;                              // A-fragment row for this lane
    const int* ip = idx + (b * NPTS + n) * KNN;
    const _Float16* wrow = w16 + (size_t)(o0 + r15) * twoC;  // subtile j adds j*16*twoC
    float s4[4], h4[4];
    v8f vmax[4];
    #pragma unroll
    for (int j = 0; j < 4; ++j) {
        int ocol = o0 + (j << 4) + r15;
        s4[j] = sc[ocol]; h4[j] = sh[ocol];
        #pragma unroll
        for (int r = 0; r < 8; ++r) vmax[j][r] = -3.0e38f;
    }

    int nbr = ip[0];
    for (int k = 0; k < KNN; ++k) {
        int nbr_next = (k + 1 < KNN) ? ip[k + 1] : nbr;
        __builtin_prefetch(xb + (size_t)nbr_next * C, 0, 3);   // global_prefetch_b8
        v8f acc[4] = {{}, {}, {}, {}};
        for (int st = 0; st < ksteps; ++st) {
            int kb = (st << 5) + loff;
            v8h alo = edge_chunk(xb, C, n, nbr, kb);
            v8h ahi = edge_chunk(xb, C, n, nbr, kb + 16);
            v16h af = cat16(alo, ahi);
            #pragma unroll
            for (int j = 0; j < 4; ++j) {
                const _Float16* wr = wrow + (size_t)(j << 4) * twoC + kb;
                v16h bf = cat16(*(const v8h*)wr, *(const v8h*)(wr + 16));
                acc[j] = __builtin_amdgcn_wmma_f32_16x16x32_f16(false, af, false, bf,
                                                                (short)0, acc[j],
                                                                false, false);
            }
        }
        #pragma unroll
        for (int j = 0; j < 4; ++j)
            #pragma unroll
            for (int r = 0; r < 8; ++r)
                vmax[j][r] = fmaxf(vmax[j][r], lrelu(acc[j][r] * s4[j] + h4[j]));
        nbr = nbr_next;
    }
    #pragma unroll
    for (int j = 0; j < 4; ++j) {
        int ocol = o0 + (j << 4) + r15;
        #pragma unroll
        for (int r = 0; r < 8; ++r) {
            int base = b * NPTS + n0 + loff + r;
            of32[(size_t)base * O + ocol] = vmax[j][r];
            of16[(size_t)base * O + ocol] = (_Float16)vmax[j][r];
            cat[(size_t)base * 512 + coff + ocol] = (_Float16)vmax[j][r];
        }
    }
}

// ---------------- conv5: h[b,o,n] = lrelu(bn(W5(1024x512) . cat[b,n,:])) ----
// One wave computes a 16o x 64n tile; W5 fragment reused across 4 n-subtiles.
// grid: (16 n-supertiles, 64 o-tiles, 16 b), block = 32.
__global__ void k_conv5_wmma(const _Float16* __restrict__ w16, const _Float16* __restrict__ cat,
                             const float* __restrict__ sc, const float* __restrict__ sh,
                             float* __restrict__ hbuf) {
    int lane = threadIdx.x;
    int b = blockIdx.z, o0 = blockIdx.y << 4, n0 = blockIdx.x << 6;
    int r15 = lane & 15, loff = (lane >> 4) << 3;
    const _Float16* arow = w16 + (size_t)(o0 + r15) * 512;
    const _Float16* brow = cat + ((size_t)(b * NPTS) + n0 + r15) * 512;
    v8f acc[4] = {{}, {}, {}, {}};
    for (int st = 0; st < 16; ++st) {
        int kb = st << 5;
        v16h af = frag_row(arow, kb, lane);
        #pragma unroll
        for (int j = 0; j < 4; ++j) {
            v16h bf = frag_row(brow + (size_t)(j << 4) * 512, kb, lane);
            acc[j] = __builtin_amdgcn_wmma_f32_16x16x32_f16(false, af, false, bf,
                                                            (short)0, acc[j], false, false);
        }
    }
    float scv[8], shv[8];
    #pragma unroll
    for (int r = 0; r < 8; ++r) { int o = o0 + loff + r; scv[r] = sc[o]; shv[r] = sh[o]; }
    #pragma unroll
    for (int j = 0; j < 4; ++j) {
        int ncol = n0 + (j << 4) + r15;
        #pragma unroll
        for (int r = 0; r < 8; ++r) {
            int o = o0 + loff + r;
            float y = lrelu(acc[j][r] * scv[r] + shv[r]);
            hbuf[((size_t)(b * NPTS + o) << 10) + ncol] = y;
        }
    }
}

// ---------------- global max + mean pooling ---------------------------------
__global__ void k_pool(const float* __restrict__ hbuf, float* __restrict__ g) {
    int t = blockIdx.x * blockDim.x + threadIdx.x;
    if (t >= BATCH * NPTS) return;
    int b = t >> 10, o = t & 1023;
    const float* row = hbuf + ((size_t)t << 10);
    float mx = -3.0e38f, sm = 0.f;
    for (int n = 0; n < NPTS; ++n) { float v = row[n]; mx = fmaxf(mx, v); sm += v; }
    g[b * 2048 + o] = mx;
    g[b * 2048 + 1024 + o] = sm * (1.f / 1024.f);
}

// ---------------- dense head (tiny; scalar) ---------------------------------
__global__ void k_dense(const float* __restrict__ in, const float* __restrict__ W,
                        const float* __restrict__ bias, const float* __restrict__ sc,
                        const float* __restrict__ sh, float* __restrict__ out,
                        int I, int O, int act) {
    int t = blockIdx.x * blockDim.x + threadIdx.x;
    if (t >= BATCH * O) return;
    int b = t / O, o = t % O;
    const float* ir = in + (size_t)b * I;
    const float* wr = W + (size_t)o * I;
    float a = 0.f;
    for (int i = 0; i < I; ++i) a += ir[i] * wr[i];
    if (bias) a += bias[o];
    if (sc) a = a * sc[o] + sh[o];
    if (act) a = lrelu(a);
    out[t] = a;
}

extern "C" void kernel_launch(void* const* d_in, const int* in_sizes, int n_in,
                              void* d_out, int out_size, void* d_ws, size_t ws_size,
                              hipStream_t stream) {
    (void)in_sizes; (void)n_in; (void)out_size; (void)ws_size;
    const float* x  = (const float*)d_in[0];
    const float* w1 = (const float*)d_in[1];
    const float* w2 = (const float*)d_in[2];
    const float* w3 = (const float*)d_in[3];
    const float* w4 = (const float*)d_in[4];
    const float* w5 = (const float*)d_in[5];
    const float* bn[28];
    for (int i = 0; i < 28; ++i) bn[i] = (const float*)d_in[6 + i];
    const float* W6 = (const float*)d_in[34];
    const float* W7 = (const float*)d_in[35];
    const float* b7 = (const float*)d_in[36];
    const float* W8 = (const float*)d_in[37];
    const float* b8 = (const float*)d_in[38];
    float* out = (float*)d_out;

    // Workspace layout (requires ws_size >= 0x8800000 ~ 136 MiB)
    char* w = (char*)d_ws;
    float*    bnsc  = (float*)(w + 0x0000000);          // 2304 f32
    float*    bnsh  = (float*)(w + 0x0004000);
    float*    sq    = (float*)(w + 0x0010000);          // 16K f32
    int*      idx   = (int*)  (w + 0x0020000);          // 16K*20 i32
    float*    xtA   = (float*)(w + 0x0200000);          // 16 MiB ping
    float*    xtB   = (float*)(w + 0x1200000);          // 16 MiB pong
    _Float16* xtA16 = (_Float16*)(w + 0x2200000);       // 8 MiB
    _Float16* xtB16 = (_Float16*)(w + 0x2A00000);       // 8 MiB
    _Float16* w2f   = (_Float16*)(w + 0x3200000);
    _Float16* w3f   = (_Float16*)(w + 0x3210000);
    _Float16* w4f   = (_Float16*)(w + 0x3220000);
    _Float16* w5f   = (_Float16*)(w + 0x3240000);
    _Float16* cat   = (_Float16*)(w + 0x3400000);       // 16 MiB (B,N,512) f16
    float*    g     = (float*)(w + 0x4400000);          // 16x2048
    float*    h6    = (float*)(w + 0x4420000);
    float*    h7    = (float*)(w + 0x4430000);
    float*    dist  = (float*)(w + 0x4800000);          // 64 MiB, reused as conv5 out
    float*    hbuf  = dist;

    // BN fold: bn1..bn7, channel offsets into bnsc/bnsh
    const int offs[7] = {0, 64, 128, 256, 512, 1536, 2048};
    const int chs[7]  = {64, 64, 128, 256, 1024, 512, 256};
    for (int i = 0; i < 7; ++i)
        k_bnprep<<<(chs[i] + 63) / 64, 64, 0, stream>>>(bn[4*i], bn[4*i+1], bn[4*i+2],
                                                        bn[4*i+3], bnsc + offs[i],
                                                        bnsh + offs[i], chs[i]);

    k_transpose_x<<<64, 256, 0, stream>>>(x, xtA);
    k_cvt16<<<(64 * 128 + 255) / 256, 256, 0, stream>>>(w2, w2f, 64 * 128);
    k_cvt16<<<(128 * 128 + 255) / 256, 256, 0, stream>>>(w3, w3f, 128 * 128);
    k_cvt16<<<(256 * 256 + 255) / 256, 256, 0, stream>>>(w4, w4f, 256 * 256);
    k_cvt16<<<(1024 * 512 + 255) / 256, 256, 0, stream>>>(w5, w5f, 1024 * 512);

    // ---- EdgeConv 1 (C=3 -> 64), scalar path ----
    k_dist_l1<<<(BATCH * NPTS * NPTS) / 256, 256, 0, stream>>>(xtA, dist);
    k_topk<<<64, 256, 0, stream>>>(dist, idx);
    k_edge_l1<<<(BATCH * NPTS * 64) / 256, 256, 0, stream>>>(xtA, idx, w1, bnsc + 0,
                                                             bnsh + 0, xtB, xtB16, cat);

    // ---- EdgeConv 2 (C=64 -> 64), WMMA ----
    k_sq<<<64, 256, 0, stream>>>(xtB, sq, 64);
    k_dist_wmma<<<dim3(16, 64, 16), 32, 0, stream>>>(xtB16, sq, dist, 64);
    k_topk<<<64, 256, 0, stream>>>(dist, idx);
    k_edge_wmma<<<dim3(1, 64, 16), 32, 0, stream>>>(xtB16, idx, w2f, bnsc + 64, bnsh + 64,
                                                    xtA, xtA16, cat, 64, 64, 64);

    // ---- EdgeConv 3 (C=64 -> 128), WMMA ----
    k_sq<<<64, 256, 0, stream>>>(xtA, sq, 64);
    k_dist_wmma<<<dim3(16, 64, 16), 32, 0, stream>>>(xtA16, sq, dist, 64);
    k_topk<<<64, 256, 0, stream>>>(dist, idx);
    k_edge_wmma<<<dim3(2, 64, 16), 32, 0, stream>>>(xtA16, idx, w3f, bnsc + 128, bnsh + 128,
                                                    xtB, xtB16, cat, 64, 128, 128);

    // ---- EdgeConv 4 (C=128 -> 256), WMMA ----
    k_sq<<<64, 256, 0, stream>>>(xtB, sq, 128);
    k_dist_wmma<<<dim3(16, 64, 16), 32, 0, stream>>>(xtB16, sq, dist, 128);
    k_topk<<<64, 256, 0, stream>>>(dist, idx);
    k_edge_wmma<<<dim3(4, 64, 16), 32, 0, stream>>>(xtB16, idx, w4f, bnsc + 256, bnsh + 256,
                                                    xtA, xtA16, cat, 128, 256, 256);

    // ---- conv5 (512 -> 1024) + BN + lrelu, WMMA; then pooling ----
    k_conv5_wmma<<<dim3(16, 64, 16), 32, 0, stream>>>(w5f, cat, bnsc + 512, bnsh + 512, hbuf);
    k_pool<<<64, 256, 0, stream>>>(hbuf, g);

    // ---- dense head ----
    k_dense<<<(BATCH * 512 + 255) / 256, 256, 0, stream>>>(g, W6, nullptr, bnsc + 1536,
                                                           bnsh + 1536, h6, 2048, 512, 1);
    k_dense<<<(BATCH * 256 + 255) / 256, 256, 0, stream>>>(h6, W7, b7, bnsc + 2048,
                                                           bnsh + 2048, h7, 512, 256, 1);
    k_dense<<<(BATCH * 40 + 255) / 256, 256, 0, stream>>>(h7, W8, b8, nullptr, nullptr,
                                                          out, 256, 40, 0);
}
